// MultiHeadAttention_58815282151425
// MI455X (gfx1250) — matrix operand order repaired and checked by reading, not compiled, not run on hardware
//
#include <hip/hip_runtime.h>
#include <hip/hip_bf16.h>

// ---------------------------------------------------------------------------
// MI455X (gfx1250) sliding-window MHA.
//   * all three matmul stages on v_wmma_f32_16x16x32_bf16 (fp32 accumulate)
//   * intermediates (qkv, Y) stored as bf16: they are only ever consumed as
//     bf16 WMMA operands, so this halves global traffic with zero extra loss
//   * attention K/V/Q tile staging via GLOBAL_LOAD_ASYNC_TO_LDS_B128
//     (ASYNCcnt path); register-copy fallback if the builtin is absent
// ---------------------------------------------------------------------------

typedef __attribute__((ext_vector_type(16))) __bf16 v16bf;
typedef __attribute__((ext_vector_type(8)))  float  v8f;
typedef __attribute__((ext_vector_type(4)))  int    v4i;

typedef __attribute__((address_space(1))) v4i g_v4i;   // global int4
typedef __attribute__((address_space(3))) v4i l_v4i;   // LDS int4

#define S_LEN   2048
#define NB      2
#define NH      8
#define DH      64
#define EMB     512
#define QKV_N   1536      // 3*E ; per-head layout: [h*192 + {Q:0..63, K:64..127, V:128..191}]
#define HALF_WIN 128
#define NEGV    (-1e30f)

__device__ __forceinline__ __bf16 tobf(float f)  { return (__bf16)f; }
__device__ __forceinline__ __bf16 tobf(__bf16 f) { return f; }
__device__ __forceinline__ int imin(int a, int b) { return a < b ? a : b; }
__device__ __forceinline__ int imax(int a, int b) { return a > b ? a : b; }

__device__ __forceinline__ void store_elem(float*  p, float v) { *p = v; }
__device__ __forceinline__ void store_elem(__bf16* p, float v) { *p = (__bf16)v; }

// ---- CDNA5 async global->LDS staging (16B per lane per issue) --------------
#define HAS_ASYNC_LDS __has_builtin(__builtin_amdgcn_global_load_async_to_lds_b128)

__device__ __forceinline__ void copy16_g2l(const __bf16* g, __bf16* l) {
#if HAS_ASYNC_LDS
    __builtin_amdgcn_global_load_async_to_lds_b128(
        (g_v4i*)g, (l_v4i*)l, /*offset=*/0, /*cpol=*/0);
#else
    *(float4*)l = *(const float4*)g;
#endif
}

__device__ __forceinline__ void wait_g2l() {
#if HAS_ASYNC_LDS
  #if __has_builtin(__builtin_amdgcn_s_wait_asynccnt)
    __builtin_amdgcn_s_wait_asynccnt(0);
  #else
    asm volatile("s_wait_asynccnt 0x0" ::: "memory");
  #endif
#endif
}

// ---------------------------------------------------------------------------
// out[m][n] = sum_k A[m][k] * W[n][k] + bias[n]
// A: fp32 or bf16; out: fp32 or bf16. M,N multiples of 64, K multiple of 32.
// 128 threads = 4 waves; 64x64 block tile; each wave owns a 16x64 strip.
// ---------------------------------------------------------------------------
template <typename AT, typename OT>
__global__ void gemm_at_wt(const AT* __restrict__ A, const float* __restrict__ W,
                           const float* __restrict__ bias, OT* __restrict__ out,
                           int M, int N, int K)
{
    __shared__ __align__(16) __bf16 As[64][32];
    __shared__ __align__(16) __bf16 Ws[64][32];

    const int tid  = threadIdx.x;
    const int lane = tid & 31;
    const int wave = tid >> 5;
    const int half = lane >> 4;       // half-wave: K-split of the fragments
    const int l16  = lane & 15;
    const int m0   = blockIdx.y * 64;
    const int n0   = blockIdx.x * 64;

    v8f acc[4] = {};

    for (int k0 = 0; k0 < K; k0 += 32) {
        for (int e = tid; e < 64 * 32; e += 128) {
            int r = e >> 5, c = e & 31;
            As[r][c] = tobf(A[(size_t)(m0 + r) * K + k0 + c]);
            Ws[r][c] = tobf(W[(size_t)(n0 + r) * K + k0 + c]);
        }
        __syncthreads();

        v16bf afrag;   // A fragment: lane=M row, half-wave splits K ranges
        #pragma unroll
        for (int i = 0; i < 16; ++i) {
            int kk = i + (half ? 16 : 0);
            afrag[i] = As[wave * 16 + l16][kk];
        }
        #pragma unroll
        for (int t = 0; t < 4; ++t) {
            v16bf bfrag;  // B fragment: lane=N col, half-wave K 0-15/16-31
            #pragma unroll
            for (int i = 0; i < 16; ++i)
                bfrag[i] = Ws[t * 16 + l16][(half ? 16 : 0) + i];
            acc[t] = __builtin_amdgcn_wmma_f32_16x16x32_bf16(
                false, afrag, false, bfrag, (short)0, acc[t], false, false);
        }
        __syncthreads();
    }

    #pragma unroll
    for (int t = 0; t < 4; ++t) {
        #pragma unroll
        for (int r = 0; r < 8; ++r) {
            int row = m0 + wave * 16 + r + (half ? 8 : 0);
            int col = n0 + t * 16 + l16;
            store_elem(out + (size_t)row * N + col, acc[t][r] + bias[col]);
        }
    }
}

// ---------------------------------------------------------------------------
// Banded attention: one wave per (b, h, 16-row query tile).
// qkv is bf16 [tok][1536]. 9 key-chunks of 32 cover the 272-wide band.
// Two-pass softmax with the band in LDS; LDS also provides the C->A layout
// transpose for P before the PV WMMAs. K/V/Q tiles staged via async-to-LDS.
// ---------------------------------------------------------------------------
__global__ void band_attn(const __bf16* __restrict__ qkv, const int* __restrict__ pad,
                          __bf16* __restrict__ Y)
{
    __shared__ __align__(16) float  sc[16][288];   // fp32 band scores
    __shared__ __align__(16) __bf16 pa[16][288];   // normalized probs
    __shared__ __align__(16) __bf16 qt[16][64];
    __shared__ __align__(16) __bf16 kt[32][64];
    __shared__ __align__(16) __bf16 vt[32][64];

    const int lane = threadIdx.x & 31;
    const int half = lane >> 4;
    const int l16  = lane & 15;
    const int q0   = blockIdx.x * 16;
    const int h    = blockIdx.y;
    const int b    = blockIdx.z;
    const size_t baseTok = (size_t)b * S_LEN;
    const __bf16* qkvh = qkv + h * 192;            // head base within a row

    // ---- stage Q tile (16x64 bf16 = 128 x 16B chunks) ----
    #pragma unroll
    for (int j = 0; j < 4; ++j) {
        int c = j * 32 + lane;
        int r = c >> 3, dc = c & 7;                // 8 chunks per 64-dim row
        copy16_g2l(qkvh + (baseTok + q0 + r) * QKV_N + dc * 8, &qt[r][dc * 8]);
    }
    wait_g2l();
    __syncthreads();

    v16bf qa[2];   // Q A-fragments over the two head-dim chunks
    #pragma unroll
    for (int c = 0; c < 2; ++c)
        #pragma unroll
        for (int i = 0; i < 16; ++i) {
            int dd = c * 32 + i + (half ? 16 : 0);
            qa[c][i] = qt[l16][dd];
        }

    int rowpadbits = 0;   // pad flags for this lane's 8 accumulator rows
    #pragma unroll
    for (int r = 0; r < 8; ++r)
        if (pad[b * S_LEN + q0 + r + half * 8]) rowpadbits |= (1 << r);

    const int kbase0 = q0 - HALF_WIN;

    // ---- pass 1: S = (Q K^T) * scale, masked, into LDS ----
    for (int cc = 0; cc < 9; ++cc) {
        int kb = kbase0 + cc * 32;
        #pragma unroll
        for (int j = 0; j < 8; ++j) {              // 32x64 bf16 = 256 x 16B
            int c = j * 32 + lane;
            int kl = c >> 3, dc = c & 7;
            int kc2 = imin(imax(kb + kl, 0), S_LEN - 1);
            copy16_g2l(qkvh + (baseTok + kc2) * QKV_N + 64 + dc * 8, &kt[kl][dc * 8]);
        }
        wait_g2l();
        __syncthreads();

        #pragma unroll
        for (int s = 0; s < 2; ++s) {
            v8f sacc = {};
            #pragma unroll
            for (int c = 0; c < 2; ++c) {
                v16bf bf;  // B(d, key): lane = key col, half-wave splits d
                #pragma unroll
                for (int i = 0; i < 16; ++i)
                    bf[i] = kt[s * 16 + l16][c * 32 + (half ? 16 : 0) + i];
                sacc = __builtin_amdgcn_wmma_f32_16x16x32_bf16(
                    false, qa[c], false, bf, (short)0, sacc, false, false);
            }
            int n_g   = kb + s * 16 + l16;
            int inS   = (n_g >= 0 && n_g < S_LEN);
            int colOK = inS ? pad[b * S_LEN + imin(imax(n_g, 0), S_LEN - 1)] : 0;
            #pragma unroll
            for (int r = 0; r < 8; ++r) {
                int m = q0 + r + half * 8;
                int dlt = m - n_g; dlt = dlt < 0 ? -dlt : dlt;
                bool v = colOK && ((rowpadbits >> r) & 1) && (dlt <= HALF_WIN);
                sc[r + half * 8][cc * 32 + s * 16 + l16] = v ? sacc[r] * 0.125f : NEGV;
            }
        }
        __syncthreads();
    }

    // ---- softmax: row = lane%16, half-waves split the 288 columns ----
    {
        int row = l16;
        int c0 = half * 144, c1 = c0 + 144;
        float mx = NEGV;
        for (int cI = c0; cI < c1; ++cI) mx = fmaxf(mx, sc[row][cI]);
        mx = fmaxf(mx, __shfl_xor(mx, 16, 32));
        float sum = 0.f;
        for (int cI = c0; cI < c1; ++cI) sum += __expf(sc[row][cI] - mx);
        sum += __shfl_xor(sum, 16, 32);
        bool  rv  = mx > -1e29f;                   // row has any valid entry
        float inv = rv ? 1.0f / sum : 0.0f;
        for (int cI = c0; cI < c1; ++cI)
            pa[row][cI] = tobf(rv ? __expf(sc[row][cI] - mx) * inv : 0.0f);
    }
    __syncthreads();

    // ---- pass 2: O = P V ----
    v8f o[4] = {};
    for (int cc = 0; cc < 9; ++cc) {
        int kb = kbase0 + cc * 32;
        #pragma unroll
        for (int j = 0; j < 8; ++j) {
            int c = j * 32 + lane;
            int kl = c >> 3, dc = c & 7;
            int kc2 = imin(imax(kb + kl, 0), S_LEN - 1);   // P==0 outside band
            copy16_g2l(qkvh + (baseTok + kc2) * QKV_N + 128 + dc * 8, &vt[kl][dc * 8]);
        }
        wait_g2l();
        __syncthreads();

        v16bf pfr;  // P as A fragment: lane=row, K = local key index in chunk
        #pragma unroll
        for (int i = 0; i < 16; ++i) {
            int kk = cc * 32 + i + (half ? 16 : 0);
            pfr[i] = pa[l16][kk];
        }
        #pragma unroll
        for (int t = 0; t < 4; ++t) {
            v16bf bf;  // B(key, dim): lane = dim col, half-wave splits keys
            #pragma unroll
            for (int i = 0; i < 16; ++i)
                bf[i] = vt[(half ? 16 : 0) + i][t * 16 + l16];
            o[t] = __builtin_amdgcn_wmma_f32_16x16x32_bf16(
                false, pfr, false, bf, (short)0, o[t], false, false);
        }
        __syncthreads();
    }

    // ---- store Y[token][h*64 + d] as bf16 ----
    #pragma unroll
    for (int t = 0; t < 4; ++t)
        #pragma unroll
        for (int r = 0; r < 8; ++r) {
            int row = q0 + r + half * 8;
            int col = h * DH + t * 16 + l16;
            Y[(baseTok + row) * EMB + col] = tobf(o[t][r]);
        }
}

// ---------------------------------------------------------------------------
extern "C" void kernel_launch(void* const* d_in, const int* in_sizes, int n_in,
                              void* d_out, int out_size, void* d_ws, size_t ws_size,
                              hipStream_t stream) {
    (void)in_sizes; (void)n_in; (void)out_size; (void)ws_size;
    const float* x    = (const float*)d_in[0];
    const int*   pad  = (const int*)  d_in[1];
    const float* Wqkv = (const float*)d_in[2];
    const float* bqkv = (const float*)d_in[3];
    const float* Wo   = (const float*)d_in[4];
    const float* bo   = (const float*)d_in[5];
    float* out = (float*)d_out;

    const int M = NB * S_LEN;                                  // 4096 tokens
    __bf16* qkv  = (__bf16*)d_ws;                              // [M,1536] bf16 (12 MB)
    __bf16* Ybuf = (__bf16*)((char*)d_ws + (size_t)M * QKV_N * sizeof(__bf16)); // [M,512] bf16 (4 MB)

    // 1) QKV projection: qkv(bf16) = x @ Wqkv^T + bqkv
    gemm_at_wt<float, __bf16>
        <<<dim3(QKV_N / 64, M / 64), 128, 0, stream>>>(x, Wqkv, bqkv, qkv, M, QKV_N, EMB);

    // 2) Sliding-window attention per (b, h, 16-row query tile)
    band_attn<<<dim3(S_LEN / 16, NH, NB), 32, 0, stream>>>(qkv, pad, Ybuf);

    // 3) Output projection: out(fp32) = Y @ Wo^T + bo
    gemm_at_wt<__bf16, float>
        <<<dim3(EMB / 64, M / 64), 128, 0, stream>>>(Ybuf, Wo, bo, out, M, EMB, EMB);
}
